// KMeans_53463752901141
// MI455X (gfx1250) — matrix operand order repaired and checked by reading, not compiled, not run on hardware
//
#include <hip/hip_runtime.h>
#include <hip/hip_bf16.h>

// KMeans assignment: out[b] = argmin_k ||X[b]-C[k]||^2
// = argmin_k (c2[k] - 2 * X[b].C[k])   (x^2 term constant per row)
// Cross term via V_WMMA_F32_16X16X32_F16 with f16 hi/lo split (3 passes),
// f32 accumulation. X tile staged in LDS (full-F reuse across all centroids).

typedef __attribute__((ext_vector_type(4)))  _Float16 v4h;
typedef __attribute__((ext_vector_type(8)))  _Float16 v8h;
typedef __attribute__((ext_vector_type(16))) _Float16 v16h;
typedef __attribute__((ext_vector_type(8)))  float    v8f;

#define F_DIM 1024
#define K_CENT 1024
#define M_TILE 64          // rows per workgroup
#define ROWH 1032          // padded row stride in halves (2064B: stride%64dw == 4 -> no bank conflicts)

__device__ __forceinline__ v16h combine16(v8h a, v8h b) {
  return __builtin_shufflevector(a, b, 0,1,2,3,4,5,6,7,8,9,10,11,12,13,14,15);
}

__device__ __forceinline__ v8f wmma_f16(v16h a, v16h b, v8f c) {
  return __builtin_amdgcn_wmma_f32_16x16x32_f16(false, a, false, b, (short)0, c, false, false);
}

// ---------------- Phase 0: centroid prep (f32 -> f16 hi/lo, c2 norms) ----------------
__global__ void KMeans_prep_kernel(const float* __restrict__ C,
                                   _Float16* __restrict__ Chi,
                                   _Float16* __restrict__ Clo,
                                   float* __restrict__ c2) {
  __shared__ float red[256];
  const int k = blockIdx.x;
  const int tid = threadIdx.x;
  const float* row = C + (size_t)k * F_DIM;
  const int i = tid * 4;                       // 256 threads * 4 = 1024 elements
  float4 v = *(const float4*)(row + i);
  _Float16 h0 = (_Float16)v.x, h1 = (_Float16)v.y, h2 = (_Float16)v.z, h3 = (_Float16)v.w;
  v4h hh = {h0, h1, h2, h3};
  v4h ll = {(_Float16)(v.x - (float)h0), (_Float16)(v.y - (float)h1),
            (_Float16)(v.z - (float)h2), (_Float16)(v.w - (float)h3)};
  *(v4h*)(Chi + (size_t)k * F_DIM + i) = hh;
  *(v4h*)(Clo + (size_t)k * F_DIM + i) = ll;
  red[tid] = v.x*v.x + v.y*v.y + v.z*v.z + v.w*v.w;
  __syncthreads();
  for (int off = 128; off > 0; off >>= 1) {
    if (tid < off) red[tid] += red[tid + off];
    __syncthreads();
  }
  if (tid == 0) c2[k] = red[0];
}

// ---------------- Phase 1: WMMA GEMM + fused argmin ----------------
__global__ __launch_bounds__(256, 1)
void KMeans_assign_kernel(const float* __restrict__ X,
                          const _Float16* __restrict__ Chi,
                          const _Float16* __restrict__ Clo,
                          const float* __restrict__ c2,
                          int* __restrict__ out) {
  __shared__ _Float16 Ahi[M_TILE * ROWH];     // 129 KB
  __shared__ _Float16 Alo[M_TILE * ROWH];     // 129 KB
  __shared__ unsigned long long best[M_TILE]; // packed {key32, idx32}

  const int tid = threadIdx.x;
  const int rowbase = blockIdx.x * M_TILE;

  // ---- Stage X tile into LDS as f16 hi/lo (read HBM exactly once) ----
  {
    const float4* xg = (const float4*)(X + (size_t)rowbase * F_DIM);
    for (int i = tid; i < M_TILE * F_DIM / 4; i += 256) {
      float4 v = xg[i];
      const int e = i * 4;
      const int m = e >> 10;          // F_DIM == 1024
      const int f = e & (F_DIM - 1);
      _Float16 h0 = (_Float16)v.x, h1 = (_Float16)v.y, h2 = (_Float16)v.z, h3 = (_Float16)v.w;
      v4h hh = {h0, h1, h2, h3};
      v4h ll = {(_Float16)(v.x - (float)h0), (_Float16)(v.y - (float)h1),
                (_Float16)(v.z - (float)h2), (_Float16)(v.w - (float)h3)};
      *(v4h*)&Ahi[m * ROWH + f] = hh;
      *(v4h*)&Alo[m * ROWH + f] = ll;
    }
    if (tid < M_TILE) best[tid] = 0xFFFFFFFFFFFFFFFFull;
  }
  __syncthreads();

  const int wave   = tid >> 5;
  const int lane   = tid & 31;
  const int mblock = wave & 3;        // 4 M-blocks of 16 rows
  const int ngroup = wave >> 2;       // 2 N-groups of 64 cols
  const int l15    = lane & 15;
  const int khalf  = lane >> 4;       // A: K-halves 0..7/16..23 vs 8..15/24..31

  const _Float16* Abase_hi = &Ahi[(mblock * 16 + l15) * ROWH];
  const _Float16* Abase_lo = &Alo[(mblock * 16 + l15) * ROWH];

  unsigned bkey[8], bidx[8];
#pragma unroll
  for (int j = 0; j < 8; ++j) { bkey[j] = 0xFFFFFFFFu; bidx[j] = 0u; }

  for (int p = 0; p < K_CENT / 128; ++p) {
    const int colbase = p * 128 + ngroup * 64 + l15;
    v8f acc[4];
#pragma unroll
    for (int t = 0; t < 4; ++t) acc[t] = (v8f){0.f,0.f,0.f,0.f,0.f,0.f,0.f,0.f};

    for (int f = 0; f < F_DIM; f += 32) {
      // A fragment (16x32 f16): lanes0-15 K {f..f+7, f+16..f+23}; lanes16-31 +8
      const v8h* ah = (const v8h*)(Abase_hi + f + khalf * 8);
      const v8h* al = (const v8h*)(Abase_lo + f + khalf * 8);
      v16h ahi = combine16(ah[0], ah[2]);   // +0 halves, +16 halves
      v16h alo = combine16(al[0], al[2]);
#pragma unroll
      for (int t = 0; t < 4; ++t) {
        const int col = colbase + t * 16;   // lane's centroid column
        // B fragment (32x16 f16): lane's column, K {f..f+7, f+16..f+23}; lanes16-31 +8
        const v8h* bh = (const v8h*)(Chi + (size_t)col * F_DIM + f + khalf * 8);
        const v8h* bl = (const v8h*)(Clo + (size_t)col * F_DIM + f + khalf * 8);
        v16h bhi = combine16(bh[0], bh[2]);
        v16h blo = combine16(bl[0], bl[2]);
        acc[t] = wmma_f16(ahi, bhi, acc[t]);   // hi*hi
        acc[t] = wmma_f16(ahi, blo, acc[t]);   // hi*lo
        acc[t] = wmma_f16(alo, bhi, acc[t]);   // lo*hi
      }
    }

    // score = c2[col] - 2*cross ; running argmin in registers
#pragma unroll
    for (int t = 0; t < 4; ++t) {
      const int col = colbase + t * 16;
      const float cc = c2[col];
#pragma unroll
      for (int j = 0; j < 8; ++j) {
        const float s = fmaf(-2.0f, acc[t][j], cc);
        const unsigned u = __float_as_uint(s);
        const unsigned key = (u & 0x80000000u) ? ~u : (u | 0x80000000u);
        if (key < bkey[j]) { bkey[j] = key; bidx[j] = (unsigned)col; }
      }
    }
  }

  // Cross-wave reduction: packed u64 min keeps smallest score, then smallest idx
#pragma unroll
  for (int j = 0; j < 8; ++j) {
    const int r = mblock * 16 + khalf * 8 + j;   // local row 0..63
    const unsigned long long pk =
        ((unsigned long long)bkey[j] << 32) | (unsigned long long)bidx[j];
    atomicMin(&best[r], pk);
  }
  __syncthreads();
  if (tid < M_TILE) out[rowbase + tid] = (int)(unsigned)(best[tid] & 0xFFFFFFFFull);
}

extern "C" void kernel_launch(void* const* d_in, const int* in_sizes, int n_in,
                              void* d_out, int out_size, void* d_ws, size_t ws_size,
                              hipStream_t stream) {
  const float* X = (const float*)d_in[0];          // [65536,1024] f32
  const float* C = (const float*)d_in[1];          // [1024,1024]  f32
  // workspace: Chi (2MB) | Clo (2MB) | c2 (4KB)
  _Float16* Chi = (_Float16*)d_ws;
  _Float16* Clo = Chi + (size_t)K_CENT * F_DIM;
  float*    c2  = (float*)(Clo + (size_t)K_CENT * F_DIM);
  int*      out = (int*)d_out;

  KMeans_prep_kernel<<<K_CENT, 256, 0, stream>>>(C, Chi, Clo, c2);
  KMeans_assign_kernel<<<65536 / M_TILE, 256, 0, stream>>>(X, Chi, Clo, c2, out);
}
